// XAIDiffNet_841813590553
// MI455X (gfx1250) — matrix operand order repaired and checked by reading, not compiled, not run on hardware
//
#include <hip/hip_runtime.h>
#include <hip/hip_bf16.h>

// ---------------------------------------------------------------------------
// DCRNN forward for MI455X (gfx1250, wave32).
// Compute-bound (~3 TFLOP, working set < L2) -> route all GEMMs through
// v_wmma_f32_16x16x32_bf16 (bf16 operands, f32 accum).
// GEMM: 256 thr / 8 waves, block tile 128x64, wave tile 32x32 (4 WMMA/chunk),
// LDS-staged K chunks of 32 with pair-packed transposed B so every fragment
// is built from aligned ds_load_b128. K loop is segment-outer / chunk-inner
// with pure pointer stepping (no per-iteration division or pointer selects).
// ---------------------------------------------------------------------------

typedef __bf16 bf16_t;
typedef __attribute__((ext_vector_type(16))) __bf16 v16bf;
typedef __attribute__((ext_vector_type(8)))  float  v8f;

#define B_  32
#define T_  12
#define PL_ 12
#define N_  1024
#define U_  64
#define L_  2
#define RTOT (N_ * B_)      // 32768 (row = n*32 + b)
#define CFEAT 128           // concat feature dim
#define KCAT 384            // [X | PfX | PbX] concatenated K

// ---- fast transcendentals: gfx1250 has V_TANH_F32 ----
__device__ __forceinline__ float fast_tanh(float x)
{
#if defined(__gfx1250__) && __has_builtin(__builtin_amdgcn_tanhf)
    return __builtin_amdgcn_tanhf(x);
#else
    return tanhf(x);
#endif
}
__device__ __forceinline__ float fast_sigmoid(float x)
{
    return 0.5f * fast_tanh(0.5f * x) + 0.5f;   // exact identity
}

// ---------------------------------------------------------------------------
// D[M x Ncols] = epi( sum_K A[M x K] * B[K x Ncols] + bias )
// A = nseg row-major segments of aSegK columns each, segment s at
// A0 + s*aSegStride (aSegStride==0 -> single segment repeated; used with nseg=1).
// ---------------------------------------------------------------------------
__global__ __launch_bounds__(256)
void gemm_bf16_wmma(const bf16_t* __restrict__ A0, size_t aSegStride,
                    int nseg, int aSegK, int lda,
                    const bf16_t* __restrict__ Bm, int ldb,
                    bf16_t* __restrict__ D, int ldd,
                    const float* __restrict__ bias,
                    int K, int epilogue)
{
    // rows padded for conflict-free 16-lane fragment reads (stride 20 dwords)
    __shared__ bf16_t       As[128][40];   // cols 0..31 used
    __shared__ unsigned int Bp[64][20];    // pair-packed (k,k+1) x n, cols 0..15 used

    const int tid  = threadIdx.x;
    const int lane = tid & 31;
    const int wave = tid >> 5;
    const int wm   = wave >> 1;            // 0..3  (32 rows each)
    const int wn   = wave & 1;             // 0..1  (32 cols each)
    const int m0   = blockIdx.x * 128;
    const int n0   = blockIdx.y * 64;

    v8f acc[2][2] = {};

    const int aLr = tid >> 1;              // 0..127 A tile row
    const int aLq = tid & 1;               // 16 bf16 half-row
    const int bLk = tid >> 3;              // 0..31  B tile k-row
    const int bLq = tid & 7;               // 8 bf16 segment

    // fragment LDS addresses (loop-invariant)
    const int mbase = wm * 32 + (lane & 15);
    const int khalf = (lane >> 4) * 8;
    const int kp    = (lane >> 4) * 4;

    const size_t bStep = (size_t)32 * ldb;
    const bf16_t* bp = Bm + (size_t)bLk * ldb + n0 + bLq * 8;
    int kg = 0;                            // global k progress (for prefetch guard)

    for (int sgi = 0; sgi < nseg; ++sgi) {
        const bf16_t* ap = A0 + (size_t)sgi * aSegStride
                              + (size_t)(m0 + aLr) * lda + aLq * 16;
        for (int kl = 0; kl < aSegK; kl += 32) {
            // --- stage A tile 128x32 (row-major, vectorized) ---
            {
                uint4 t4a = *(const uint4*)(ap);
                uint4 t4b = *(const uint4*)(ap + 8);
                *(uint4*)&As[aLr][aLq * 16]     = t4a;
                *(uint4*)&As[aLr][aLq * 16 + 8] = t4b;
            }
            // --- stage B tile 32x64: vector load, transpose+pair-pack to LDS ---
            {
                union { uint4 t4; unsigned short s[8]; } bb;
                bb.t4 = *(const uint4*)bp;
#pragma unroll
                for (int j = 0; j < 8; ++j) {
                    unsigned short* dst = (unsigned short*)&Bp[bLq * 8 + j][bLk >> 1];
                    dst[bLk & 1] = bb.s[j];
                }
            }
            if (kg + 32 < K)   // gfx1250 global_prefetch_b8 for next B chunk
                __builtin_prefetch(bp + bStep, 0, 0);
            __syncthreads();

            // --- fragments: all aligned 16B LDS loads ---
            union { unsigned int u[8]; v16bf v; } afr[2], bfr[2];
#pragma unroll
            for (int s = 0; s < 2; ++s) {
                const bf16_t* arow = &As[mbase + s * 16][0];
                *(uint4*)&afr[s].u[0] = *(const uint4*)(arow + khalf);
                *(uint4*)&afr[s].u[4] = *(const uint4*)(arow + khalf + 16);
            }
#pragma unroll
            for (int t = 0; t < 2; ++t) {
                const unsigned int* brow = &Bp[wn * 32 + t * 16 + (lane & 15)][0];
                *(uint4*)&bfr[t].u[0] = *(const uint4*)(brow + kp);
                *(uint4*)&bfr[t].u[4] = *(const uint4*)(brow + kp + 8);
            }

#pragma unroll
            for (int s = 0; s < 2; ++s)
#pragma unroll
                for (int t = 0; t < 2; ++t)
                    acc[s][t] = __builtin_amdgcn_wmma_f32_16x16x32_bf16(
                        false, afr[s].v, false, bfr[t].v, (short)0, acc[s][t], false, false);
            __syncthreads();

            ap += 32;
            bp += bStep;
            kg += 32;
        }
    }

    // --- epilogue + store (C/D: m = i + (lane>>4)*8, n = lane&15) ---
    const int mloc = (lane >> 4) * 8;
    const int nc   = lane & 15;
#pragma unroll
    for (int s = 0; s < 2; ++s)
#pragma unroll
        for (int t = 0; t < 2; ++t)
#pragma unroll
            for (int i = 0; i < 8; ++i) {
                const int m_g = m0 + wm * 32 + s * 16 + mloc + i;
                const int n_g = n0 + wn * 32 + t * 16 + nc;
                float val = acc[s][t][i] + (bias ? bias[n_g] : 0.0f);
                if (epilogue == 1)      val = fast_sigmoid(val);
                else if (epilogue == 2) val = fast_tanh(val);
                D[(size_t)m_g * ldd + n_g] = (bf16_t)val;
            }
}

// --------------------------- support matrices ------------------------------
__global__ __launch_bounds__(256)
void support_sums(const float* __restrict__ adj, float* __restrict__ dsum)
{
    __shared__ float s1[256];
    __shared__ float s2[256];
    const int m = blockIdx.x, tid = threadIdx.x;
    float a1 = 0.f, a2 = 0.f;
    for (int n = tid; n < N_; n += 256) {
        const float I = (n == m) ? 1.0f : 0.0f;
        a1 += adj[(size_t)m * N_ + n] + I;   // row sum  -> d_out
        a2 += adj[(size_t)n * N_ + m] + I;   // col sum  -> d_in
    }
    s1[tid] = a1; s2[tid] = a2;
    __syncthreads();
    for (int s = 128; s > 0; s >>= 1) {
        if (tid < s) { s1[tid] += s1[tid + s]; s2[tid] += s2[tid + s]; }
        __syncthreads();
    }
    if (tid == 0) { dsum[m] = s1[0]; dsum[N_ + m] = s2[0]; }
}

__global__ __launch_bounds__(256)
void build_supports(const float* __restrict__ adj, const float* __restrict__ dsum,
                    bf16_t* __restrict__ Pf, bf16_t* __restrict__ Pb)
{
    const int idx = blockIdx.x * 256 + threadIdx.x;     // over N*N
    const int m = idx >> 10, n = idx & 1023;
    const float I  = (m == n) ? 1.0f : 0.0f;
    const float e  = adj[(size_t)m * N_ + n] + I;
    const float eT = adj[(size_t)n * N_ + m] + I;
    const float df = dsum[m], db = dsum[N_ + m];
    Pf[idx] = (bf16_t)(df > 0.f ? e / df : 0.f);
    Pb[idx] = (bf16_t)(db > 0.f ? eT / db : 0.f);
}

// ------------------------ weight packing (bf16) ----------------------------
// Wcat[k][o], k in [0,384): k<128 -> Wf[l,0]+Wb[l,0]; <256 -> Wf[l,1]; else Wb[l,1]
__global__ __launch_bounds__(256)
void pack_wcat(const float* __restrict__ Wf, const float* __restrict__ Wb,
               bf16_t* __restrict__ dst, int od, int layer)
{
    const int idx = blockIdx.x * 256 + threadIdx.x;     // over 384*od
    const int k = idx / od, o = idx - k * od;
    float v;
    if (k < 128)
        v = Wf[((size_t)(layer * 2 + 0) * 128 + k) * od + o] +
            Wb[((size_t)(layer * 2 + 0) * 128 + k) * od + o];
    else if (k < 256)
        v = Wf[((size_t)(layer * 2 + 1) * 128 + (k - 128)) * od + o];
    else
        v = Wb[((size_t)(layer * 2 + 1) * 128 + (k - 256)) * od + o];
    dst[idx] = (bf16_t)v;
}

// ------------------------------ elementwise --------------------------------
__global__ __launch_bounds__(256)
void zero_f32(float* __restrict__ ptr, int n)
{
    const int idx = blockIdx.x * 256 + threadIdx.x;
    if (idx < n) ptr[idx] = 0.0f;
}

// emb[r][u] = src[b][n] * in_w[u] + in_b[u]   (src==null -> zeros, go0)
__global__ __launch_bounds__(256)
void embed_kernel(const float* __restrict__ src, int bstride,
                  const float* __restrict__ in_w, const float* __restrict__ in_b,
                  bf16_t* __restrict__ emb)
{
    const int idx = blockIdx.x * 256 + threadIdx.x;     // over RTOT*64
    const int r = idx >> 6, u = idx & 63;
    const int b = r & 31, n = r >> 5;
    const float xv = src ? src[(size_t)b * bstride + n] : 0.0f;
    emb[idx] = (bf16_t)(xv * in_w[u] + in_b[u]);
}

// F[r][0:64] = x_in ; F[r][64:128] = h (optionally * r-gate)
__global__ __launch_bounds__(256)
void concat_kernel(const bf16_t* __restrict__ xbf, const float* __restrict__ xf32,
                   const float* __restrict__ h, const bf16_t* __restrict__ rgate,
                   bf16_t* __restrict__ F)
{
    const int idx = blockIdx.x * 256 + threadIdx.x;     // over RTOT*128
    const int r = idx >> 7, d = idx & 127;
    float v;
    if (d < 64) {
        v = xbf ? (float)xbf[(size_t)r * 64 + d] : xf32[(size_t)r * 64 + d];
    } else {
        float hv = h[(size_t)r * 64 + (d - 64)];
        if (rgate) hv *= (float)rgate[(size_t)r * 128 + (d - 64)];   // r = ru[:, :64]
        v = hv;
    }
    F[idx] = (bf16_t)v;
}

// h = (1-u)*h + u*c ;  u = (u1+u2)/2, c = (c1+c2)/2
__global__ __launch_bounds__(256)
void combine_kernel(const bf16_t* __restrict__ ru1, const bf16_t* __restrict__ ru2,
                    const bf16_t* __restrict__ c1,  const bf16_t* __restrict__ c2,
                    float* __restrict__ h)
{
    const int idx = blockIdx.x * 256 + threadIdx.x;     // over RTOT*64
    const int r = idx >> 6, u = idx & 63;
    const float uu = 0.5f * ((float)ru1[(size_t)r * 128 + 64 + u] +
                             (float)ru2[(size_t)r * 128 + 64 + u]);
    const float cc = 0.5f * ((float)c1[idx] + (float)c2[idx]);
    h[idx] = (1.0f - uu) * h[idx] + uu * cc;
}

// out[b][pl][n] = h_last[r] . out_w + out_b
__global__ __launch_bounds__(256)
void outproj_kernel(const float* __restrict__ h, const float* __restrict__ out_w,
                    const float* __restrict__ out_b, float* __restrict__ out, int pl)
{
    const int r = blockIdx.x * 256 + threadIdx.x;       // over RTOT
    const int b = r & 31, n = r >> 5;
    float s = out_b[0];
#pragma unroll
    for (int u = 0; u < 64; ++u) s += h[(size_t)r * 64 + u] * out_w[u];
    out[((size_t)b * PL_ + pl) * N_ + n] = s;
}

// ---------------------------------------------------------------------------
extern "C" void kernel_launch(void* const* d_in, const int* in_sizes, int n_in,
                              void* d_out, int out_size, void* d_ws, size_t ws_size,
                              hipStream_t stream)
{
    const float* x    = (const float*)d_in[0];
    const float* adj  = (const float*)d_in[1];
    const float* in_w = (const float*)d_in[2];
    const float* in_b = (const float*)d_in[3];
    const float* outw = (const float*)d_in[4];
    const float* outb = (const float*)d_in[5];
    float* out = (float*)d_out;

    // params dict insertion order: stage{enc,dec} x path{p1,p2} x part{gate,cand} x {Wf,Wb,b}
    auto widx = [](int stage, int path, int part, int which) {
        return 6 + stage * 12 + path * 6 + part * 3 + which;
    };

    // ---- workspace carve (bump allocator, 256B aligned) ----
    char* p = (char*)d_ws;
    auto carve = [&](size_t bytes) -> char* {
        char* r = p;
        p += (bytes + 255) & ~(size_t)255;
        return r;
    };
    bf16_t* Pf   = (bf16_t*)carve((size_t)N_ * N_ * 2);
    bf16_t* Pb   = (bf16_t*)carve((size_t)N_ * N_ * 2);
    float*  dsum = (float*)carve((size_t)2 * N_ * 4);
    bf16_t* Wg[2][2][2];   // [stage][path][layer] 384x128
    bf16_t* Wc[2][2][2];   // [stage][path][layer] 384x64
    for (int s = 0; s < 2; ++s)
        for (int q = 0; q < 2; ++q)
            for (int l = 0; l < L_; ++l) {
                Wg[s][q][l] = (bf16_t*)carve((size_t)KCAT * 128 * 2);
                Wc[s][q][l] = (bf16_t*)carve((size_t)KCAT * 64 * 2);
            }
    bf16_t* emb = (bf16_t*)carve((size_t)RTOT * 64 * 2);
    // Fx, Ff, Fb carved contiguously: weight-GEMM A segments are Fx + s*segStride
    bf16_t* Fx  = (bf16_t*)carve((size_t)RTOT * CFEAT * 2);
    bf16_t* Ff  = (bf16_t*)carve((size_t)RTOT * CFEAT * 2);
    bf16_t* Fb  = (bf16_t*)carve((size_t)RTOT * CFEAT * 2);
    bf16_t* ru1 = (bf16_t*)carve((size_t)RTOT * 128 * 2);
    bf16_t* ru2 = (bf16_t*)carve((size_t)RTOT * 128 * 2);
    bf16_t* c1  = (bf16_t*)carve((size_t)RTOT * 64 * 2);
    bf16_t* c2  = (bf16_t*)carve((size_t)RTOT * 64 * 2);
    float*  h   = (float*)carve((size_t)L_ * RTOT * 64 * 4);

    const size_t segStride = (size_t)(Ff - Fx);   // == RTOT*CFEAT elements

    // ---- one-time per-call setup ----
    support_sums<<<N_, 256, 0, stream>>>(adj, dsum);
    build_supports<<<(N_ * N_) / 256, 256, 0, stream>>>(adj, dsum, Pf, Pb);
    for (int s = 0; s < 2; ++s)
        for (int q = 0; q < 2; ++q)
            for (int l = 0; l < L_; ++l) {
                pack_wcat<<<(KCAT * 128) / 256, 256, 0, stream>>>(
                    (const float*)d_in[widx(s, q, 0, 0)], (const float*)d_in[widx(s, q, 0, 1)],
                    Wg[s][q][l], 128, l);
                pack_wcat<<<(KCAT * 64) / 256, 256, 0, stream>>>(
                    (const float*)d_in[widx(s, q, 1, 0)], (const float*)d_in[widx(s, q, 1, 1)],
                    Wc[s][q][l], 64, l);
            }
    const int hElems = L_ * RTOT * 64;
    zero_f32<<<(hElems + 255) / 256, 256, 0, stream>>>(h, hElems);

    const int NCOLS = B_ * CFEAT;                 // 4096 columns for propagation
    const dim3 gProp(N_ / 128, NCOLS / 64);       // 8 x 64
    const dim3 gGate(RTOT / 128, 128 / 64);       // 256 x 2
    const dim3 gCand(RTOT / 128, 64 / 64);        // 256 x 1
    const int nConcat = (RTOT * CFEAT) / 256;
    const int nComb   = (RTOT * 64) / 256;

    // ---- time loop: encoder then decoder ----
    for (int stage = 0; stage < 2; ++stage) {
        const int steps = (stage == 0) ? T_ : PL_;
        for (int t = 0; t < steps; ++t) {
            const float* src;
            int bstride;
            if (stage == 0) { src = x + (size_t)t * N_; bstride = T_ * N_; }
            else            { src = (t == 0) ? nullptr : out + (size_t)(t - 1) * N_; bstride = PL_ * N_; }
            embed_kernel<<<(RTOT * 64) / 256, 256, 0, stream>>>(src, bstride, in_w, in_b, emb);

            for (int l = 0; l < L_; ++l) {
                const bf16_t* xbf = (l == 0) ? emb : nullptr;
                const float*  xf  = (l == 0) ? nullptr : (h + (size_t)(l - 1) * RTOT * 64);
                float* hl = h + (size_t)l * RTOT * 64;
                const float* gb1 = (const float*)d_in[widx(stage, 0, 0, 2)] + l * 128;
                const float* gb2 = (const float*)d_in[widx(stage, 1, 0, 2)] + l * 128;
                const float* cb1 = (const float*)d_in[widx(stage, 0, 1, 2)] + l * 64;
                const float* cb2 = (const float*)d_in[widx(stage, 1, 1, 2)] + l * 64;

                // gate: xh -> prop (shared by p1/p2) -> two fused-sigmoid GEMMs
                concat_kernel<<<nConcat, 256, 0, stream>>>(xbf, xf, hl, nullptr, Fx);
                gemm_bf16_wmma<<<gProp, 256, 0, stream>>>(Pf, 0, 1, N_, N_, Fx, NCOLS,
                                                          Ff, NCOLS, nullptr, N_, 0);
                gemm_bf16_wmma<<<gProp, 256, 0, stream>>>(Pb, 0, 1, N_, N_, Fx, NCOLS,
                                                          Fb, NCOLS, nullptr, N_, 0);
                gemm_bf16_wmma<<<gGate, 256, 0, stream>>>(Fx, segStride, 3, CFEAT, CFEAT,
                                                          Wg[stage][0][l], 128, ru1, 128,
                                                          gb1, KCAT, 1);
                gemm_bf16_wmma<<<gGate, 256, 0, stream>>>(Fx, segStride, 3, CFEAT, CFEAT,
                                                          Wg[stage][1][l], 128, ru2, 128,
                                                          gb2, KCAT, 1);

                // candidate path 1: [x | r1*h] -> prop -> fused-tanh GEMM
                concat_kernel<<<nConcat, 256, 0, stream>>>(xbf, xf, hl, ru1, Fx);
                gemm_bf16_wmma<<<gProp, 256, 0, stream>>>(Pf, 0, 1, N_, N_, Fx, NCOLS,
                                                          Ff, NCOLS, nullptr, N_, 0);
                gemm_bf16_wmma<<<gProp, 256, 0, stream>>>(Pb, 0, 1, N_, N_, Fx, NCOLS,
                                                          Fb, NCOLS, nullptr, N_, 0);
                gemm_bf16_wmma<<<gCand, 256, 0, stream>>>(Fx, segStride, 3, CFEAT, CFEAT,
                                                          Wc[stage][0][l], 64, c1, 64,
                                                          cb1, KCAT, 2);

                // candidate path 2
                concat_kernel<<<nConcat, 256, 0, stream>>>(xbf, xf, hl, ru2, Fx);
                gemm_bf16_wmma<<<gProp, 256, 0, stream>>>(Pf, 0, 1, N_, N_, Fx, NCOLS,
                                                          Ff, NCOLS, nullptr, N_, 0);
                gemm_bf16_wmma<<<gProp, 256, 0, stream>>>(Pb, 0, 1, N_, N_, Fx, NCOLS,
                                                          Fb, NCOLS, nullptr, N_, 0);
                gemm_bf16_wmma<<<gCand, 256, 0, stream>>>(Fx, segStride, 3, CFEAT, CFEAT,
                                                          Wc[stage][1][l], 64, c2, 64,
                                                          cb2, KCAT, 2);

                // GRU combine (in-place h update feeds next layer / next step)
                combine_kernel<<<nComb, 256, 0, stream>>>(ru1, ru2, c1, c2, hl);
            }
            if (stage == 1)
                outproj_kernel<<<RTOT / 256, 256, 0, stream>>>(
                    h + (size_t)(L_ - 1) * RTOT * 64, outw, outb, out, t);
        }
    }
}
